// Attention_72060961292580
// MI455X (gfx1250) — matrix operand order repaired and checked by reading, not compiled, not run on hardware
//
#include <hip/hip_runtime.h>
#include <hip/hip_bf16.h>
#include <math.h>

typedef __attribute__((ext_vector_type(16))) __bf16 v16bf;
typedef __attribute__((ext_vector_type(2)))  __bf16 v2bf;
typedef __attribute__((ext_vector_type(8)))  float  v8f;

#define BATCH 2
#define SEQ   2048
#define TKV   2048
#define NH    32
#define NKV   8
#define REP   4
#define HD    128
#define BM    128            // query rows per workgroup (8 waves x 16)
#define BN    64             // kv tile
#define WAVES 8
#define KSTRIDE (HD + 8)     // bf16 elems; 272B row, 16B aligned
#define VSTRIDE (BN + 8)     // 144B row
#define PSTRIDE (BN + 8)

// ---- 16-lane-group reductions on the VALU via DPP16 (no LDS-pipe traffic) ----
// xor1 = quad_perm[1,0,3,2] (0xB1), xor2 = quad_perm[2,3,0,1] (0x4E),
// then row_half_mirror (lane^7 == xor4 once 4-groups uniform, 0x141),
// then row_mirror      (lane^15 == xor8 once 8-groups uniform, 0x140).
#if __has_builtin(__builtin_amdgcn_update_dpp)
template <int CTRL>
__device__ __forceinline__ float dpp_mov(float x) {
    union { float f; int i; } u, r;
    u.f = x;
    r.i = __builtin_amdgcn_update_dpp(0, u.i, CTRL, 0xF, 0xF, true);
    return r.f;
}
__device__ __forceinline__ float red16_max(float x) {
    x = fmaxf(x, dpp_mov<0xB1>(x));
    x = fmaxf(x, dpp_mov<0x4E>(x));
    x = fmaxf(x, dpp_mov<0x141>(x));
    x = fmaxf(x, dpp_mov<0x140>(x));
    return x;
}
__device__ __forceinline__ float red16_sum(float x) {
    x += dpp_mov<0xB1>(x);
    x += dpp_mov<0x4E>(x);
    x += dpp_mov<0x141>(x);
    x += dpp_mov<0x140>(x);
    return x;
}
#else
__device__ __forceinline__ float red16_max(float x) {
    for (int off = 8; off >= 1; off >>= 1) x = fmaxf(x, __shfl_xor(x, off, 32));
    return x;
}
__device__ __forceinline__ float red16_sum(float x) {
    for (int off = 8; off >= 1; off >>= 1) x += __shfl_xor(x, off, 32);
    return x;
}
#endif

__global__ __launch_bounds__(256, 1)
void fa_fwd(const float* __restrict__ q, const float* __restrict__ k,
            const float* __restrict__ v, const float* __restrict__ mask,
            float* __restrict__ out)
{
    __shared__ __bf16 lK[BN][KSTRIDE];          // [key][d]
    __shared__ __bf16 lVt[HD][VSTRIDE];         // [d][key] (transposed)
    __shared__ __bf16 lP[WAVES][16][PSTRIDE];   // per-wave P staging

    const int tid  = threadIdx.x;
    const int wave = tid >> 5;
    const int lane = tid & 31;
    const int l15  = lane & 15;
    const int hi   = lane >> 4;      // half-wave select
    const int hi8  = hi * 8;
    const int hi16 = hi * 16;

    const int h  = blockIdx.y;       // query head
    const int b  = blockIdx.z;
    const int g  = h / REP;          // kv head (GQA)
    const int m0 = blockIdx.x * BM + wave * 16;

    const float scale = 0.08838834764831845f;   // 1/sqrt(128)

    // ---- Q: 16x128 rows -> 4 bf16 A-fragments (K=32 each), pre-scaled ----
    v16bf qf[4];
    {
        const float* qrow = q + ((size_t)b * SEQ + (size_t)(m0 + l15)) * (NH * HD)
                              + (size_t)h * HD;
        #pragma unroll
        for (int f = 0; f < 4; ++f) {
            const int d0 = 32 * f + hi8;
            v16bf a;
            #pragma unroll
            for (int e = 0; e < 8; ++e) a[e]     = (__bf16)(qrow[d0 + e] * scale);
            #pragma unroll
            for (int e = 0; e < 8; ++e) a[8 + e] = (__bf16)(qrow[d0 + 16 + e] * scale);
            qf[f] = a;
        }
    }

    v8f zero = {};
    v8f o[8];                 // O accumulator: 16 rows x 128 cols, fp32
    #pragma unroll
    for (int dt = 0; dt < 8; ++dt) o[dt] = zero;
    float mrun[8], lrun[8];
    #pragma unroll
    for (int i = 0; i < 8; ++i) { mrun[i] = -INFINITY; lrun[i] = 0.f; }

    const size_t kv_row = (size_t)NKV * HD;     // 1024 floats

    // cooperative loader: thread owns rows nbase+8c (c=0..7), cols dp..dp+3.
    // all chunk deltas are compile-time immediates off a single base pointer.
    const int nbase = tid >> 5;                 // 0..7
    const int dp    = (tid & 31) << 2;          // 0..124
    const float* kit = k + (size_t)b * TKV * kv_row + (size_t)g * HD
                         + (size_t)nbase * kv_row + dp;
    const float* vit = v + (size_t)b * TKV * kv_row + (size_t)g * HD
                         + (size_t)nbase * kv_row + dp;
    // per-lane mask base: rows (m0+hi8+i), cols (n0 + 16t + l15)
    const float* mit = mask + (size_t)(m0 + hi8) * TKV + l15;

    // ---- prologue: fetch tile 0 into registers (double buffer stage) ----
    float4 kreg[8], vreg[8];
    float  mreg[4][8];
    #pragma unroll
    for (int c = 0; c < 8; ++c) {
        kreg[c] = *(const float4*)(kit + (size_t)(8 * c) * kv_row);
        vreg[c] = *(const float4*)(vit + (size_t)(8 * c) * kv_row);
    }
    #pragma unroll
    for (int t = 0; t < 4; ++t)
        #pragma unroll
        for (int i = 0; i < 8; ++i)
            mreg[t][i] = mit[(size_t)i * TKV + 16 * t];

    for (int n0 = 0; n0 < TKV; n0 += BN) {
        __syncthreads();   // all waves done reading the previous LDS tile
        // ---- publish current tile: regs -> LDS (fp32 -> bf16, V transposed) ----
        #pragma unroll
        for (int c = 0; c < 8; ++c) {
            const int n = nbase + 8 * c;
            const v2bf k01 = { (__bf16)kreg[c].x, (__bf16)kreg[c].y };
            const v2bf k23 = { (__bf16)kreg[c].z, (__bf16)kreg[c].w };
            *(v2bf*)&lK[n][dp]     = k01;
            *(v2bf*)&lK[n][dp + 2] = k23;
            lVt[dp + 0][n] = (__bf16)vreg[c].x; lVt[dp + 1][n] = (__bf16)vreg[c].y;
            lVt[dp + 2][n] = (__bf16)vreg[c].z; lVt[dp + 3][n] = (__bf16)vreg[c].w;
        }
        __syncthreads();

        // ---- issue next tile's K/V global loads; latency hides under WMMAs ----
        if (n0 + BN < TKV) {   // uniform branch
            const float* kn = kit + (size_t)(n0 + BN) * kv_row;
            const float* vn = vit + (size_t)(n0 + BN) * kv_row;
            #pragma unroll
            for (int c = 0; c < 8; ++c) {
                kreg[c] = *(const float4*)(kn + (size_t)(8 * c) * kv_row);
                vreg[c] = *(const float4*)(vn + (size_t)(8 * c) * kv_row);
            }
            if (n0 + 2 * BN < TKV) {   // warm L2 two tiles ahead
                #pragma unroll
                for (int c = 0; c < 8; ++c) {
                    __builtin_prefetch(kn + (size_t)(BN + 8 * c) * kv_row, 0, 1);
                    __builtin_prefetch(vn + (size_t)(BN + 8 * c) * kv_row, 0, 1);
                }
            }
        }

        // ---- S = Q * K^T : 4 column subtiles, 4 K-chunks each ----
        v8f st[4];
        #pragma unroll
        for (int t = 0; t < 4; ++t) {
            v8f c = zero;
            #pragma unroll
            for (int f = 0; f < 4; ++f) {
                const __bf16* kp = &lK[16 * t + l15][32 * f + hi8];
                v16bf bf;
                #pragma unroll
                for (int e = 0; e < 16; ++e) bf[e] = kp[e + (e & 8)];
                c = __builtin_amdgcn_wmma_f32_16x16x32_bf16(
                        false, qf[f], false, bf, (short)0, c, false, false);
            }
            st[t] = c;
        }

        // ---- additive mask: consume prefetched registers (no memory stall) ----
        #pragma unroll
        for (int t = 0; t < 4; ++t)
            #pragma unroll
            for (int i = 0; i < 8; ++i)
                st[t][i] += mreg[t][i];

        // ---- refill mask prefetch for the NEXT tile (consumed next iter) ----
        if (n0 + BN < TKV) {   // uniform branch
            const float* mn = mit + n0 + BN;
            #pragma unroll
            for (int t = 0; t < 4; ++t)
                #pragma unroll
                for (int i = 0; i < 8; ++i)
                    mreg[t][i] = mn[(size_t)i * TKV + 16 * t];
        }

        // ---- online softmax (per row = accumulator element x half-wave) ----
        #pragma unroll
        for (int i = 0; i < 8; ++i) {
            float rmax = fmaxf(fmaxf(st[0][i], st[1][i]), fmaxf(st[2][i], st[3][i]));
            rmax = red16_max(rmax);
            const float mnew = fmaxf(mrun[i], rmax);
            const float corr = __expf(mrun[i] - mnew);
            mrun[i] = mnew;
            lrun[i] *= corr;
            #pragma unroll
            for (int dt = 0; dt < 8; ++dt) o[dt][i] *= corr;
            float rsum = 0.f;
            #pragma unroll
            for (int t = 0; t < 4; ++t) {
                const float p = __expf(st[t][i] - mnew);
                st[t][i] = p;
                rsum += p;
            }
            lrun[i] += red16_sum(rsum);
        }

        // ---- stage P (C-layout) -> LDS -> reload in A-layout (bf16) ----
        #pragma unroll
        for (int t = 0; t < 4; ++t)
            #pragma unroll
            for (int i = 0; i < 8; ++i)
                lP[wave][i + hi8][16 * t + l15] = (__bf16)st[t][i];
        // same-wave RAW on LDS: compiler inserts s_wait_dscnt
        v16bf pa[2];
        #pragma unroll
        for (int f = 0; f < 2; ++f) {
            const __bf16* pp = &lP[wave][l15][32 * f + hi8];
            v16bf a;
            #pragma unroll
            for (int e = 0; e < 8; ++e) a[e]     = pp[e];
            #pragma unroll
            for (int e = 0; e < 8; ++e) a[8 + e] = pp[16 + e];
            pa[f] = a;
        }

        // ---- O += P * V ----
        #pragma unroll
        for (int dt = 0; dt < 8; ++dt) {
            #pragma unroll
            for (int f = 0; f < 2; ++f) {
                const __bf16* vp = &lVt[16 * dt + l15][32 * f + hi8];
                v16bf bf;
                #pragma unroll
                for (int e = 0; e < 16; ++e) bf[e] = vp[e + (e & 8)];
                o[dt] = __builtin_amdgcn_wmma_f32_16x16x32_bf16(
                            false, pa[f], false, bf, (short)0, o[dt], false, false);
            }
        }
    }

    // ---- epilogue: normalize and store fp32 ----
    float* obase = out + ((size_t)b * SEQ + (size_t)m0) * (NH * HD) + (size_t)h * HD;
    #pragma unroll
    for (int i = 0; i < 8; ++i) {
        const float inv = 1.f / lrun[i];
        float* orow = obase + (size_t)(i + hi8) * (NH * HD);
        #pragma unroll
        for (int dt = 0; dt < 8; ++dt)
            orow[16 * dt + l15] = o[dt][i] * inv;
    }
}

extern "C" void kernel_launch(void* const* d_in, const int* in_sizes, int n_in,
                              void* d_out, int out_size, void* d_ws, size_t ws_size,
                              hipStream_t stream) {
    (void)in_sizes; (void)n_in; (void)d_ws; (void)ws_size; (void)out_size;
    const float* q    = (const float*)d_in[0];
    const float* k    = (const float*)d_in[1];
    const float* v    = (const float*)d_in[2];
    // d_in[3] = start_pos (scalar 0, unused by the math)
    const float* mask = (const float*)d_in[4];
    float* out = (float*)d_out;

    dim3 grid(SEQ / BM, NH, BATCH);   // (16, 32, 2) = 1024 workgroups
    fa_fwd<<<grid, 256, 0, stream>>>(q, k, v, mask, out);
}
